// VanillaCharRNN_1288490189444
// MI455X (gfx1250) — matrix operand order repaired and checked by reading, not compiled, not run on hardware
//
#include <hip/hip_runtime.h>
#include <hip/hip_bf16.h>

// VanillaCharRNN on MI455X (gfx1250, wave32, WMMA).
// Strategy:
//   1) Precompute X0[t,b,:] = embedding[ids[b,t]] @ w_xh0^T + b_h0 for ALL t
//      (fused gather + bf16 WMMA GEMM, fully parallel, off the critical path).
//   2) Sequential scan: per step two small dual-source bf16 WMMA GEMMs
//      h0' = tanh(X0[t] + h0 Whh0^T);  h1' = tanh(h0' Wxh1^T + h1 Whh1^T + b).
//      h1' is also stashed (bf16) into H1all.
//   3) One big batched GEMM logits = H1all @ w_hy^T + b_y with the [b][t][v]
//      permuted store directly into d_out.
// bf16 inputs / f32 accumulate: V_WMMA_F32_16X16X32_BF16 (8x the K-depth of
// the f32 16x16x4 WMMA). Weights/activations are small-magnitude and
// tanh-bounded, well inside bf16 range.

#define B_ 128
#define T_ 1024
#define E_ 512
#define H_ 1024
#define V_ 512

typedef __attribute__((ext_vector_type(16))) __bf16 v16bf;
typedef __attribute__((ext_vector_type(8)))  __bf16 v8bf;
typedef __attribute__((ext_vector_type(8)))  float  v8f;
typedef __attribute__((ext_vector_type(4)))  float  v4f;

// ---- WMMA operand loaders (per CDNA5 ISA 7.12.2 16-bit layouts) -------------
// A tile 16x32 (MxK) bf16: lane m=lane&15; lanes 0-15 hold K{0..7,16..23},
// lanes 16-31 hold K{8..15,24..31}.  row_k0 = &A[m][k0] + (lane>>4)*8.
static __device__ inline v16bf ld_a_tile(const __bf16* row_k0) {
  v8bf lo = *(const v8bf*)(row_k0);
  v8bf hi = *(const v8bf*)(row_k0 + 16);
  return __builtin_shufflevector(lo, hi, 0,1,2,3,4,5,6,7,8,9,10,11,12,13,14,15);
}

// B tile 32x16 (KxN) = (W^T) tile with W row-major [N,K]: lane n=lane&15 holds
// W[n][k0 + (lane>>4)*8 .. +7] and W[n][k0 + 16 + (lane>>4)*8 .. +7].
static __device__ inline v16bf ld_b_tile(const __bf16* __restrict__ W, int ldW,
                                         int nbase, int k0, int lane) {
  const int n = nbase + (lane & 15);
  const __bf16* p = W + (size_t)n * ldW + k0 + (lane >> 4) * 8;
  v8bf lo = *(const v8bf*)(p);
  v8bf hi = *(const v8bf*)(p + 16);
  return __builtin_shufflevector(lo, hi, 0,1,2,3,4,5,6,7,8,9,10,11,12,13,14,15);
}

static __device__ inline v8f wmma_bf16(v16bf a, v16bf b, v8f c) {
  return __builtin_amdgcn_wmma_f32_16x16x32_bf16(
      /*neg_a=*/false, a, /*neg_b=*/false, b,
      /*c_mod=*/(short)0, c, /*reuse_a=*/false, /*reuse_b=*/false);
}

// ---- Dual-source GEMM: D = act(C0 + A1*W1^T + A2*W2^T + bias) ---------------
// Block = 64 threads (2 waves). Wave computes a 16x32 output tile:
// rows [m0,m0+16), cols [n0,n0+32). Grid = (N/32, M/32).
template <bool TANH>
__global__ __launch_bounds__(64) void rnn_gemm_kernel(
    const __bf16* __restrict__ A1, int K1, const __bf16* __restrict__ W1,
    const __bf16* __restrict__ A2, int K2, const __bf16* __restrict__ W2,
    const __bf16* __restrict__ C0, const float* __restrict__ bias, int N,
    __bf16* __restrict__ outB, __bf16* __restrict__ outB2,
    float* __restrict__ outF) {
  const int lane = threadIdx.x & 31;
  const int wave = threadIdx.x >> 5;
  const int n0   = blockIdx.x * 32;
  const int m0   = blockIdx.y * 32 + wave * 16;
  const int nlo  = lane & 15;
  const int half = lane >> 4;

  v8f c0, c1;
  const float b0v = bias ? bias[n0 + nlo] : 0.f;
  const float b1v = bias ? bias[n0 + 16 + nlo] : 0.f;
#pragma unroll
  for (int j = 0; j < 8; ++j) {
    const int m = m0 + j + 8 * half;  // C/D layout: VGPR j <-> row m0+j+8*half
    c0[j] = b0v + (C0 ? (float)C0[(size_t)m * N + n0 + nlo] : 0.f);
    c1[j] = b1v + (C0 ? (float)C0[(size_t)m * N + n0 + 16 + nlo] : 0.f);
  }

  {
    const __bf16* arow = A1 + (size_t)(m0 + nlo) * K1 + half * 8;
    for (int k0 = 0; k0 < K1; k0 += 32) {
      __builtin_prefetch(arow + k0 + 64, 0, 1);
      v16bf a  = ld_a_tile(arow + k0);
      v16bf bb0 = ld_b_tile(W1, K1, n0,      k0, lane);
      v16bf bb1 = ld_b_tile(W1, K1, n0 + 16, k0, lane);
      c0 = wmma_bf16(a, bb0, c0);
      c1 = wmma_bf16(a, bb1, c1);
    }
  }
  if (A2) {
    const __bf16* arow = A2 + (size_t)(m0 + nlo) * K2 + half * 8;
    for (int k0 = 0; k0 < K2; k0 += 32) {
      __builtin_prefetch(arow + k0 + 64, 0, 1);
      v16bf a  = ld_a_tile(arow + k0);
      v16bf bb0 = ld_b_tile(W2, K2, n0,      k0, lane);
      v16bf bb1 = ld_b_tile(W2, K2, n0 + 16, k0, lane);
      c0 = wmma_bf16(a, bb0, c0);
      c1 = wmma_bf16(a, bb1, c1);
    }
  }

#pragma unroll
  for (int j = 0; j < 8; ++j) {
    const int m = m0 + j + 8 * half;
    float v0 = c0[j], v1 = c1[j];
    if (TANH) { v0 = tanhf(v0); v1 = tanhf(v1); }
    const size_t o0 = (size_t)m * N + n0 + nlo;
    const size_t o1 = o0 + 16;
    if (outB)  { outB[o0]  = (__bf16)v0; outB[o1]  = (__bf16)v1; }
    if (outB2) { outB2[o0] = (__bf16)v0; outB2[o1] = (__bf16)v1; }
    if (outF)  { outF[o0]  = v0;         outF[o1]  = v1; }
  }
}

// ---- Fused embedding-gather + input GEMM ------------------------------------
// Rows r = t*128 + b over all (t,b): X0[r,:] = bf16(emb[ids[b,t]] @ Wx^T + b_h0)
__global__ __launch_bounds__(64) void embed_x0_kernel(
    const int* __restrict__ ids, const float* __restrict__ emb,
    const __bf16* __restrict__ Wx, const float* __restrict__ bh0,
    __bf16* __restrict__ X0) {
  const int lane = threadIdx.x & 31;
  const int wave = threadIdx.x >> 5;
  const int n0   = blockIdx.x * 32;
  const int m0   = blockIdx.y * 32 + wave * 16;
  const int nlo  = lane & 15;
  const int half = lane >> 4;

  const int r  = m0 + nlo;            // this lane's A-row
  const int t  = r >> 7;
  const int b  = r & 127;
  const int id = ids[b * T_ + t];
  const float* arow = emb + (size_t)id * E_ + half * 8;

  v8f c0, c1;
  const float b0v = bh0[n0 + nlo];
  const float b1v = bh0[n0 + 16 + nlo];
#pragma unroll
  for (int j = 0; j < 8; ++j) { c0[j] = b0v; c1[j] = b1v; }

  for (int k0 = 0; k0 < E_; k0 += 32) {
    const float* p = arow + k0;
    v4f f0 = *(const v4f*)(p);       // K k0+off+0..3
    v4f f1 = *(const v4f*)(p + 4);   // K k0+off+4..7
    v4f f2 = *(const v4f*)(p + 16);  // K k0+off+16..19
    v4f f3 = *(const v4f*)(p + 20);  // K k0+off+20..23
    v16bf a;
#pragma unroll
    for (int i = 0; i < 4; ++i) {
      a[i]      = (__bf16)f0[i];
      a[4 + i]  = (__bf16)f1[i];
      a[8 + i]  = (__bf16)f2[i];
      a[12 + i] = (__bf16)f3[i];
    }
    v16bf bb0 = ld_b_tile(Wx, E_, n0,      k0, lane);
    v16bf bb1 = ld_b_tile(Wx, E_, n0 + 16, k0, lane);
    c0 = wmma_bf16(a, bb0, c0);
    c1 = wmma_bf16(a, bb1, c1);
  }

#pragma unroll
  for (int j = 0; j < 8; ++j) {
    const int m = m0 + j + 8 * half;
    X0[(size_t)m * H_ + n0 + nlo]      = (__bf16)c0[j];
    X0[(size_t)m * H_ + n0 + 16 + nlo] = (__bf16)c1[j];
  }
}

// ---- Batched output GEMM with permuted [b][t][v] store ----------------------
__global__ __launch_bounds__(64) void logits_kernel(
    const __bf16* __restrict__ H1all, const __bf16* __restrict__ Wy,
    const float* __restrict__ by, float* __restrict__ out) {
  const int lane = threadIdx.x & 31;
  const int wave = threadIdx.x >> 5;
  const int n0   = blockIdx.x * 32;
  const int m0   = blockIdx.y * 32 + wave * 16;
  const int nlo  = lane & 15;
  const int half = lane >> 4;

  v8f c0, c1;
  const float b0v = by[n0 + nlo];
  const float b1v = by[n0 + 16 + nlo];
#pragma unroll
  for (int j = 0; j < 8; ++j) { c0[j] = b0v; c1[j] = b1v; }

  const __bf16* arow = H1all + (size_t)(m0 + nlo) * H_ + half * 8;
  for (int k0 = 0; k0 < H_; k0 += 32) {
    __builtin_prefetch(arow + k0 + 64, 0, 1);
    v16bf a  = ld_a_tile(arow + k0);
    v16bf bb0 = ld_b_tile(Wy, H_, n0,      k0, lane);
    v16bf bb1 = ld_b_tile(Wy, H_, n0 + 16, k0, lane);
    c0 = wmma_bf16(a, bb0, c0);
    c1 = wmma_bf16(a, bb1, c1);
  }

#pragma unroll
  for (int j = 0; j < 8; ++j) {
    const int r = m0 + j + 8 * half;              // r = t*128 + b
    const size_t o = ((size_t)(r & 127) * T_ + (r >> 7)) * V_;  // [b][t][*]
    out[o + n0 + nlo]      = c0[j];
    out[o + n0 + 16 + nlo] = c1[j];
  }
}

// ---- Small helpers ----------------------------------------------------------
__global__ void cvt_bf16_kernel(const float* __restrict__ s,
                                __bf16* __restrict__ d, int n) {
  const int i = blockIdx.x * 256 + threadIdx.x;
  if (i < n) d[i] = (__bf16)s[i];
}

__global__ void zero_bf16_kernel(__bf16* __restrict__ d, int n) {
  const int i = blockIdx.x * 256 + threadIdx.x;
  if (i < n) d[i] = (__bf16)0.f;
}

// ---- Host orchestration -----------------------------------------------------
extern "C" void kernel_launch(void* const* d_in, const int* in_sizes, int n_in,
                              void* d_out, int out_size, void* d_ws,
                              size_t ws_size, hipStream_t stream) {
  const int*   ids   = (const int*)d_in[0];
  const float* emb   = (const float*)d_in[1];
  const float* w_xh0 = (const float*)d_in[2];
  const float* w_hh0 = (const float*)d_in[3];
  const float* b_h0  = (const float*)d_in[4];
  const float* w_xh1 = (const float*)d_in[5];
  const float* w_hh1 = (const float*)d_in[6];
  const float* b_h1  = (const float*)d_in[7];
  const float* w_hy  = (const float*)d_in[8];
  const float* b_y   = (const float*)d_in[9];

  float* out_logits = (float*)d_out;
  float* out_h0 = out_logits + (size_t)B_ * T_ * V_;
  float* out_h1 = out_h0 + (size_t)B_ * H_;

  // Workspace carve (256B aligned). Total ~522 MB:
  //   bf16 weights ~8MB, h-state ping-pong 1MB, X0 256MB, H1all 256MB.
  char* wp = (char*)d_ws;
  auto carve = [&](size_t bytes) -> char* {
    char* p = wp;
    wp += (bytes + 255) & ~(size_t)255;
    return p;
  };
  __bf16* wb_xh0 = (__bf16*)carve((size_t)H_ * E_ * 2);
  __bf16* wb_hh0 = (__bf16*)carve((size_t)H_ * H_ * 2);
  __bf16* wb_xh1 = (__bf16*)carve((size_t)H_ * H_ * 2);
  __bf16* wb_hh1 = (__bf16*)carve((size_t)H_ * H_ * 2);
  __bf16* wb_hy  = (__bf16*)carve((size_t)V_ * H_ * 2);
  __bf16* h0buf  = (__bf16*)carve((size_t)2 * B_ * H_ * 2);  // ping-pong
  __bf16* h1buf  = (__bf16*)carve((size_t)2 * B_ * H_ * 2);  // ping-pong
  __bf16* X0     = (__bf16*)carve((size_t)T_ * B_ * H_ * 2); // 256 MB
  __bf16* H1all  = (__bf16*)carve((size_t)T_ * B_ * H_ * 2); // 256 MB

  auto cvt = [&](const float* s, __bf16* d, int n) {
    cvt_bf16_kernel<<<dim3((n + 255) / 256), dim3(256), 0, stream>>>(s, d, n);
  };
  cvt(w_xh0, wb_xh0, H_ * E_);
  cvt(w_hh0, wb_hh0, H_ * H_);
  cvt(w_xh1, wb_xh1, H_ * H_);
  cvt(w_hh1, wb_hh1, H_ * H_);
  cvt(w_hy,  wb_hy,  V_ * H_);
  zero_bf16_kernel<<<dim3((2 * B_ * H_ + 255) / 256), dim3(256), 0, stream>>>(
      h0buf, 2 * B_ * H_);
  zero_bf16_kernel<<<dim3((2 * B_ * H_ + 255) / 256), dim3(256), 0, stream>>>(
      h1buf, 2 * B_ * H_);

  // Stage 1: all input-side GEMMs at once (M = T*B = 131072 rows).
  embed_x0_kernel<<<dim3(H_ / 32, (T_ * B_) / 32), dim3(64), 0, stream>>>(
      ids, emb, wb_xh0, b_h0, X0);

  // Stage 2: sequential scan, 2 small GEMM launches per step.
  int cur = 0;
  for (int t = 0; t < T_; ++t) {
    const int nxt = cur ^ 1;
    const bool last = (t == T_ - 1);
    rnn_gemm_kernel<true><<<dim3(H_ / 32, B_ / 32), dim3(64), 0, stream>>>(
        h0buf + (size_t)cur * B_ * H_, H_, wb_hh0,
        nullptr, 0, nullptr,
        X0 + (size_t)t * B_ * H_, nullptr, H_,
        h0buf + (size_t)nxt * B_ * H_, nullptr, last ? out_h0 : nullptr);
    rnn_gemm_kernel<true><<<dim3(H_ / 32, B_ / 32), dim3(64), 0, stream>>>(
        h0buf + (size_t)nxt * B_ * H_, H_, wb_xh1,
        h1buf + (size_t)cur * B_ * H_, H_, wb_hh1,
        nullptr, b_h1, H_,
        h1buf + (size_t)nxt * B_ * H_, H1all + (size_t)t * B_ * H_,
        last ? out_h1 : nullptr);
    cur = nxt;
  }

  // Stage 3: batched output projection for all (t,b) with [b][t][v] store.
  logits_kernel<<<dim3(V_ / 32, (T_ * B_) / 32), dim3(64), 0, stream>>>(
      H1all, wb_hy, b_y, out_logits);

  (void)in_sizes; (void)n_in; (void)out_size; (void)ws_size;
}
